// SeqGGNN_59210419143216
// MI455X (gfx1250) — matrix operand order repaired and checked by reading, not compiled, not run on hardware
//
#include <hip/hip_runtime.h>
#include <hip/hip_bf16.h>

// ---------- types ----------
typedef __attribute__((ext_vector_type(16))) __bf16 v16bf;
typedef __attribute__((ext_vector_type(8)))  float  v8f;
typedef __attribute__((ext_vector_type(4)))  unsigned int u32x4;
typedef __attribute__((ext_vector_type(2)))  unsigned int u32x2;
typedef __attribute__((ext_vector_type(4)))  float        f32x4;

union BF16Tile {                 // 32 bytes: one WMMA A/B operand per lane
    v16bf v;
    u32x4 q[2];
    unsigned short us[16];
};

// round-to-nearest-even f32 -> bf16 (inputs are finite)
__device__ __forceinline__ unsigned short f2bf(float f) {
    unsigned int u = __float_as_uint(f);
    unsigned int r = u + 0x7FFFu + ((u >> 16) & 1u);
    return (unsigned short)(r >> 16);
}

// problem constants
#define VV 32000
#define DD 256
#define BB 256
#define LL 256
#define NLAYER 10
#define NROWS (BB * LL)          // 65536

// ---------- kernel 1: transpose+convert used relation weights to bf16 ----------
// WT[mat][e][d] = bf16( Wrel[layer, sel, d, e] ),  mat = layer*2 + m, sel = m==0 ? 3 : 1
__global__ __launch_bounds__(256) void k_prep(const float* __restrict__ Wrel,
                                              unsigned short* __restrict__ WT) {
    int idx = blockIdx.x * 256 + threadIdx.x;          // 20*65536 total
    int mat = idx >> 16;
    int rem = idx & 65535;
    int e = rem >> 8;
    int d = rem & 255;
    int layer = mat >> 1;
    int sel = (mat & 1) ? 1 : 3;
    float w = Wrel[((size_t)(layer * 24 + sel) << 16) + ((size_t)d << 8) + e];
    WT[((size_t)mat << 16) + ((size_t)e << 8) + d] = f2bf(w);
}

// ---------- kernel 2: embedding gather, f32 -> bf16 ----------
__global__ __launch_bounds__(256) void k_embed(const int* __restrict__ x,
                                               const float* __restrict__ emb,
                                               unsigned short* __restrict__ H0) {
    int t = threadIdx.x;
    int r = blockIdx.x * 4 + (t >> 6);                 // row 0..65535
    int c = (t & 63) * 4;                              // 4 floats per thread
    int xi = x[r];
    f32x4 f = *(const f32x4*)(emb + ((size_t)xi << 8) + c);
    u32x2 p;
    p.x = (unsigned int)f2bf(f.x) | ((unsigned int)f2bf(f.y) << 16);
    p.y = (unsigned int)f2bf(f.z) | ((unsigned int)f2bf(f.w) << 16);
    *(u32x2*)(H0 + ((size_t)r << 8) + c) = p;
}

// ---------- kernel 3: one RGCN layer ----------
// Hout = relu( (Hin @ W3 + shift(Hin) @ W1) / deg ),  bf16 in / f32 acc / bf16 out.
// Per wave: 32 rows (2 M-subtiles) x 64 cols (4 N-subtiles) so every B-tile feeds
// two WMMAs. Block = 8 waves = 64 rows x 256 cols. EXEC stays all-ones: the
// first-in-sequence mask is a branchless select on the loaded A_prev operand.
__global__ __launch_bounds__(256) void k_layer(const unsigned short* __restrict__ Hin,
                                               unsigned short* __restrict__ Hout,
                                               const unsigned short* __restrict__ WT,
                                               int layer) {
    const int t      = threadIdx.x;
    const int lane   = t & 31;
    const int wave   = t >> 5;
    const int laneLo = lane & 15;
    const bool hi    = lane >= 16;

    const int mBase   = blockIdx.x * 64 + (wave >> 2) * 32; // 32 rows per wave
    const int colBase = (wave & 3) * 64;                    // 4 N-subtiles per wave

    const unsigned short* W3 = WT + ((size_t)(layer * 2) << 16); // WT[n][k]
    const unsigned short* W1 = W3 + (1u << 16);

    const int  row0 = mBase + laneLo;                 // A rows (both lane halves)
    const int  row1 = row0 + 16;
    const bool f0   = ((row0 & (LL - 1)) == 0);       // l == 0 -> no prev, deg = 1
    const bool f1   = ((row1 & (LL - 1)) == 0);
    const unsigned short* aS0 = Hin + ((size_t)row0 << 8);
    const unsigned short* aS1 = Hin + ((size_t)row1 << 8);
    // clamp to a valid row so the load is unconditional; data is selected to 0 below
    const unsigned short* aP0 = Hin + ((size_t)(f0 ? row0 : row0 - 1) << 8);
    const unsigned short* aP1 = Hin + ((size_t)(f1 ? row1 : row1 - 1) << 8);

    v8f acc[2][4];
#pragma unroll
    for (int m = 0; m < 2; ++m)
#pragma unroll
        for (int n = 0; n < 4; ++n) acc[m][n] = (v8f)(0.0f);

    const u32x4 zq = (u32x4)(0u);

    for (int kk = 0; kk < DD; kk += 32) {
        // A layout (16-bit, 16x32): lanes 0-15 hold K[kk..kk+7] + K[kk+16..kk+23],
        // lanes 16-31 hold K[kk+8..kk+15] + K[kk+24..kk+31].
        const int o1 = kk + (hi ? 8 : 0);
        BF16Tile as0, as1, ap0, ap1;
        as0.q[0] = *(const u32x4*)(aS0 + o1);
        as0.q[1] = *(const u32x4*)(aS0 + o1 + 16);
        as1.q[0] = *(const u32x4*)(aS1 + o1);
        as1.q[1] = *(const u32x4*)(aS1 + o1 + 16);
        {
            u32x4 p0 = *(const u32x4*)(aP0 + o1);
            u32x4 p1 = *(const u32x4*)(aP0 + o1 + 16);
            ap0.q[0] = f0 ? zq : p0;                  // branchless: v_cndmask
            ap0.q[1] = f0 ? zq : p1;
        }
        {
            u32x4 p0 = *(const u32x4*)(aP1 + o1);
            u32x4 p1 = *(const u32x4*)(aP1 + o1 + 16);
            ap1.q[0] = f1 ? zq : p0;
            ap1.q[1] = f1 ? zq : p1;
        }
        // B layout (16-bit, 32x16): lanes 0-15 hold K[kk..kk+7] + K[kk+16..kk+23] of column laneLo,
        // lanes 16-31 hold K[kk+8..kk+15] + K[kk+24..kk+31]; from WT[n][k].
        const int ob = kk + (hi ? 8 : 0);
#pragma unroll
        for (int nt = 0; nt < 4; ++nt) {
            const int col = colBase + nt * 16 + laneLo;
            const unsigned short* w3p = W3 + ((size_t)col << 8) + ob;
            const unsigned short* w1p = W1 + ((size_t)col << 8) + ob;
            BF16Tile b3, b1;
            b3.q[0] = *(const u32x4*)(w3p);
            b3.q[1] = *(const u32x4*)(w3p + 16);
            b1.q[0] = *(const u32x4*)(w1p);
            b1.q[1] = *(const u32x4*)(w1p + 16);
            acc[0][nt] = __builtin_amdgcn_wmma_f32_16x16x32_bf16(
                false, as0.v, false, b3.v, (short)0, acc[0][nt], false, false);
            acc[0][nt] = __builtin_amdgcn_wmma_f32_16x16x32_bf16(
                false, ap0.v, false, b1.v, (short)0, acc[0][nt], false, false);
            acc[1][nt] = __builtin_amdgcn_wmma_f32_16x16x32_bf16(
                false, as1.v, false, b3.v, (short)0, acc[1][nt], false, false);
            acc[1][nt] = __builtin_amdgcn_wmma_f32_16x16x32_bf16(
                false, ap1.v, false, b1.v, (short)0, acc[1][nt], false, false);
        }
    }

    // C/D layout: lane = N + 16*(M>=8), VGPR r = M%8
#pragma unroll
    for (int ms = 0; ms < 2; ++ms) {
#pragma unroll
        for (int nt = 0; nt < 4; ++nt) {
            const int col = colBase + nt * 16 + laneLo;
#pragma unroll
            for (int rr = 0; rr < 8; ++rr) {
                const int orow = mBase + ms * 16 + rr + (hi ? 8 : 0);
                const float scale = ((orow & (LL - 1)) == 0) ? 1.0f : 0.5f;
                float v = acc[ms][nt][rr] * scale;
                v = v > 0.0f ? v : 0.0f;
                Hout[((size_t)orow << 8) + col] = f2bf(v);
            }
        }
    }
}

// ---------- kernel 4: out = last @ Wout.T + bout ----------
__global__ __launch_bounds__(256) void k_final(const unsigned short* __restrict__ H,
                                               const float* __restrict__ Wout,
                                               const float* __restrict__ bout,
                                               float* __restrict__ out) {
    const int t      = threadIdx.x;
    const int lane   = t & 31;
    const int wave   = t >> 5;
    const int laneLo = lane & 15;
    const bool hi    = lane >= 16;

    const int mBase = (blockIdx.x / 250) * 16;              // batch tile
    const int nBase = (blockIdx.x % 250) * 128 + wave * 16; // vocab tile per wave

    const int brow = mBase + laneLo;                        // batch index (A row)
    const unsigned short* aPtr = H + (((size_t)brow << 8) + (LL - 1)) * DD;

    const int col = nBase + laneLo;                         // vocab column (B col)
    const float* wPtr = Wout + ((size_t)col << 8);          // Wout[col][k], contiguous in k

    v8f acc = (v8f)(0.0f);
    for (int kk = 0; kk < DD; kk += 32) {
        const int o1 = kk + (hi ? 8 : 0);
        BF16Tile a;
        a.q[0] = *(const u32x4*)(aPtr + o1);
        a.q[1] = *(const u32x4*)(aPtr + o1 + 16);

        const int ob = kk + (hi ? 8 : 0);
        const f32x4 f0 = *(const f32x4*)(wPtr + ob);
        const f32x4 f1 = *(const f32x4*)(wPtr + ob + 4);
        const f32x4 f2 = *(const f32x4*)(wPtr + ob + 16);
        const f32x4 f3 = *(const f32x4*)(wPtr + ob + 20);
        BF16Tile b;
        b.us[0]  = f2bf(f0.x); b.us[1]  = f2bf(f0.y); b.us[2]  = f2bf(f0.z); b.us[3]  = f2bf(f0.w);
        b.us[4]  = f2bf(f1.x); b.us[5]  = f2bf(f1.y); b.us[6]  = f2bf(f1.z); b.us[7]  = f2bf(f1.w);
        b.us[8]  = f2bf(f2.x); b.us[9]  = f2bf(f2.y); b.us[10] = f2bf(f2.z); b.us[11] = f2bf(f2.w);
        b.us[12] = f2bf(f3.x); b.us[13] = f2bf(f3.y); b.us[14] = f2bf(f3.z); b.us[15] = f2bf(f3.w);

        acc = __builtin_amdgcn_wmma_f32_16x16x32_bf16(
            false, a.v, false, b.v, (short)0, acc, false, false);
    }

    const float bias = bout[col];
#pragma unroll
    for (int rr = 0; rr < 8; ++rr) {
        const int orow = mBase + rr + (hi ? 8 : 0);
        out[(size_t)orow * VV + col] = acc[rr] + bias;
    }
}

// ---------- host launcher ----------
extern "C" void kernel_launch(void* const* d_in, const int* in_sizes, int n_in,
                              void* d_out, int out_size, void* d_ws, size_t ws_size,
                              hipStream_t stream) {
    const int*   x    = (const int*)d_in[0];
    const float* emb  = (const float*)d_in[1];
    const float* Wrel = (const float*)d_in[2];
    const float* Wout = (const float*)d_in[3];
    const float* bout = (const float*)d_in[4];
    float* out = (float*)d_out;

    // workspace layout (bf16 = unsigned short):
    //   H0: 65536*256, H1: 65536*256, WT: 20*256*256  -> ~70 MB total
    unsigned short* H0 = (unsigned short*)d_ws;
    unsigned short* H1 = H0 + (size_t)NROWS * DD;
    unsigned short* WT = H1 + (size_t)NROWS * DD;

    k_prep <<<(NLAYER * 2 * DD * DD) / 256, 256, 0, stream>>>(Wrel, WT);
    k_embed<<<NROWS / 4,                    256, 0, stream>>>(x, emb, H0);

    unsigned short* hin  = H0;
    unsigned short* hout = H1;
    for (int layer = 0; layer < NLAYER; ++layer) {
        k_layer<<<NROWS / 64, 256, 0, stream>>>(hin, hout, WT, layer);
        unsigned short* tmp = hin; hin = hout; hout = tmp;
    }

    k_final<<<(BB / 16) * (VV / 128), 256, 0, stream>>>(hin, Wout, bout, out);
}
